// QuoteClassificationBase_66683662238423
// MI455X (gfx1250) — matrix operand order repaired and checked by reading, compiled, not run on hardware
//
#include <hip/hip_runtime.h>
#include <hip/hip_bf16.h>
#include <cstdint>

#define NEG_SLOPE 0.2f

typedef __attribute__((ext_vector_type(16))) __bf16 v16bf;
typedef __attribute__((ext_vector_type(8)))  float  v8f;

union BFrag {
    int4  q[2];
    int   i[8];
    v16bf v;
};

// native f32 -> bf16 pair (lowered to v_cvt_pk_bf16_f32 when available)
static __device__ __forceinline__ int packbf2(float lo, float hi) {
    union { __bf16 h[2]; int i; } u;
    u.h[0] = (__bf16)lo;
    u.h[1] = (__bf16)hi;
    return u.i;
}

// order-preserving float -> uint key (for atomicMax-based segment max)
static __device__ __forceinline__ uint32_t fkey(float f) {
    uint32_t u = __float_as_uint(f);
    return (u & 0x80000000u) ? ~u : (u | 0x80000000u);
}
static __device__ __forceinline__ float funkey(uint32_t k) {
    uint32_t b = (k & 0x80000000u) ? (k ^ 0x80000000u) : ~k;
    return __uint_as_float(b);
}

// ---------------------------------------------------------------------------
// Pack W[K, ncolsW] (f32 row-major) into WMMA bf16 B-fragment layout:
// Bp[((ntile*ksteps + ks)*32 + lane)*8 + v] = packed bf16 pair (K = ks*32 +
// (lane>>4)*16 + 2v + {0,1}, col = ntile*16 + (lane&15)).
// ---------------------------------------------------------------------------
__global__ void pack_b_kernel(const float* __restrict__ W, int K, int ncolsW,
                              int* __restrict__ Bp, int ntileOffset, int ksteps) {
    int idx = blockIdx.x * blockDim.x + threadIdx.x;
    int total = (ncolsW >> 4) * ksteps * 256;
    if (idx >= total) return;
    int v    = idx & 7;
    int lane = (idx >> 3) & 31;
    int ks   = (idx >> 8) % ksteps;
    int nt   = (idx >> 8) / ksteps;
    int col  = nt * 16 + (lane & 15);
    int k    = ks * 32 + (lane >> 4) * 16 + 2 * v;
    float lo = W[(size_t)k * ncolsW + col];
    float hi = W[(size_t)(k + 1) * ncolsW + col];
    Bp[(((size_t)(ntileOffset + nt) * ksteps + ks) * 32 + lane) * 8 + v] = packbf2(lo, hi);
}

// ---------------------------------------------------------------------------
// C[M, NT*16] = A[M, K] (f32, converted to bf16 in-register) @ Bp (packed).
// One wave per 16-row M-tile, NT n-tiles per wave. 8 waves / block.
// ---------------------------------------------------------------------------
template <int NT>
__global__ void __launch_bounds__(256)
gemm_bf16_kernel(const float* __restrict__ A, const int* __restrict__ Bp,
                 float* __restrict__ C, int M, int K, int lda, int ldc) {
    const int lane  = threadIdx.x & 31;
    const int wave  = threadIdx.x >> 5;
    const int mtile = blockIdx.x * 8 + wave;
    const int mbase = mtile * 16;
    if (mbase >= M) return;                 // wave-uniform exit: EXEC stays full
    const int laneM  = lane & 15;
    const int laneHi = lane >> 4;

    int row = mbase + laneM;
    if (row >= M) row = M - 1;              // clamp (duplicate loads, stores guarded)
    const float* ap0 = A + (size_t)row * lda + laneHi * 16;

    v8f acc[NT] = {};
    const int ksteps = K >> 5;

    for (int ks = 0; ks < ksteps; ++ks) {
        const float4* af = (const float4*)(ap0 + ks * 32);
        float4 a0 = af[0], a1 = af[1];      // K = base+0..7
        float4 a2 = af[2], a3 = af[3];      // K = base+8..15
        // speculative prefetch of the next k-step's A data (dropped if OOB)
        __builtin_prefetch(ap0 + (ks + 1) * 32, 0, 1);
        BFrag Af;
        Af.i[0] = packbf2(a0.x, a0.y); Af.i[1] = packbf2(a0.z, a0.w);
        Af.i[2] = packbf2(a1.x, a1.y); Af.i[3] = packbf2(a1.z, a1.w);
        Af.i[4] = packbf2(a2.x, a2.y); Af.i[5] = packbf2(a2.z, a2.w);
        Af.i[6] = packbf2(a3.x, a3.y); Af.i[7] = packbf2(a3.z, a3.w);
#pragma unroll
        for (int t = 0; t < NT; ++t) {
            BFrag Bf;
            const int4* bf = (const int4*)(Bp + ((size_t)(t * ksteps + ks) * 32 + lane) * 8);
            Bf.q[0] = bf[0];
            Bf.q[1] = bf[1];
            acc[t] = __builtin_amdgcn_wmma_f32_16x16x32_bf16(
                false, Af.v, false, Bf.v, (short)0, acc[t], false, false);
        }
    }

#pragma unroll
    for (int t = 0; t < NT; ++t) {
#pragma unroll
        for (int j = 0; j < 8; ++j) {
            int r = mbase + laneHi * 8 + j;
            if (r < M) C[(size_t)r * ldc + t * 16 + laneM] = acc[t][j];
        }
    }
}

// ---------------------------------------------------------------------------
// Init helpers
// ---------------------------------------------------------------------------
__global__ void init_u32_kernel(uint32_t* p, int n, uint32_t v) {
    int i = blockIdx.x * blockDim.x + threadIdx.x;
    if (i < n) p[i] = v;
}
__global__ void init_f32_kernel(float* p, int n, float v) {
    int i = blockIdx.x * blockDim.x + threadIdx.x;
    if (i < n) p[i] = v;
}
__global__ void init_bias_kernel(float* p, int n, int d, const float* __restrict__ b) {
    int i = blockIdx.x * blockDim.x + threadIdx.x;
    if (i < n * d) p[i] = b[i % d];
}

// ---------------------------------------------------------------------------
// Edge pass A: score[e,h] = <leakyrelu(xl[src]+xr[dst]), att[h]>; segment max
// via uint-key atomicMax. Self-loop edges are e in [E, E+N).
// ---------------------------------------------------------------------------
template <int H, int C>
__global__ void edge_score_kernel(const float* __restrict__ xl, const float* __restrict__ xr,
                                  int ldx, const float* __restrict__ att,
                                  const int* __restrict__ src, const int* __restrict__ dst,
                                  int E, int N,
                                  float* __restrict__ score, uint32_t* __restrict__ smax) {
    int e = blockIdx.x * blockDim.x + threadIdx.x;
    if (e >= E + N) return;
    int s = (e < E) ? src[e] : (e - E);
    int d = (e < E) ? dst[e] : (e - E);
    const float* pl = xl + (size_t)s * ldx;
    const float* pr = xr + (size_t)d * ldx;
#pragma unroll
    for (int h = 0; h < H; ++h) {
        float sc = 0.f;
#pragma unroll
        for (int c = 0; c < C; c += 4) {
            float4 u = *(const float4*)(pl + h * C + c);
            float4 v = *(const float4*)(pr + h * C + c);
            float4 a = *(const float4*)(att + h * C + c);
            float m0 = u.x + v.x, m1 = u.y + v.y, m2 = u.z + v.z, m3 = u.w + v.w;
            m0 = m0 > 0.f ? m0 : NEG_SLOPE * m0;
            m1 = m1 > 0.f ? m1 : NEG_SLOPE * m1;
            m2 = m2 > 0.f ? m2 : NEG_SLOPE * m2;
            m3 = m3 > 0.f ? m3 : NEG_SLOPE * m3;
            sc += m0 * a.x + m1 * a.y + m2 * a.z + m3 * a.w;
        }
        score[(size_t)e * H + h] = sc;
        atomicMax(&smax[(size_t)d * H + h], fkey(sc));
    }
}

// Edge pass B: score <- exp(score - smax[dst]); denom[dst] += exp
template <int H>
__global__ void edge_exp_kernel(const int* __restrict__ dst, int E, int N,
                                float* __restrict__ score,
                                const uint32_t* __restrict__ smax,
                                float* __restrict__ denom) {
    int e = blockIdx.x * blockDim.x + threadIdx.x;
    if (e >= E + N) return;
    int d = (e < E) ? dst[e] : (e - E);
#pragma unroll
    for (int h = 0; h < H; ++h) {
        float sm = funkey(smax[(size_t)d * H + h]);
        float ex = __expf(score[(size_t)e * H + h] - sm);
        score[(size_t)e * H + h] = ex;
        atomicAdd(&denom[(size_t)d * H + h], ex);
    }
}

// Edge pass C: out[dst] += (ex/denom[dst]) * xl[src]
template <int H, int C>
__global__ void edge_aggr_kernel(const float* __restrict__ xl, int ldx,
                                 const int* __restrict__ src, const int* __restrict__ dst,
                                 int E, int N,
                                 const float* __restrict__ score,
                                 const float* __restrict__ denom,
                                 float* __restrict__ out, int ldo) {
    int e = blockIdx.x * blockDim.x + threadIdx.x;
    if (e >= E + N) return;
    int s = (e < E) ? src[e] : (e - E);
    int d = (e < E) ? dst[e] : (e - E);
    const float* pl = xl + (size_t)s * ldx;
    float* po = out + (size_t)d * ldo;
#pragma unroll
    for (int h = 0; h < H; ++h) {
        float alpha = score[(size_t)e * H + h] / denom[(size_t)d * H + h];
#pragma unroll
        for (int c = 0; c < C; c += 4) {
            float4 u = *(const float4*)(pl + h * C + c);
            atomicAdd(po + h * C + c + 0, alpha * u.x);
            atomicAdd(po + h * C + c + 1, alpha * u.y);
            atomicAdd(po + h * C + c + 2, alpha * u.z);
            atomicAdd(po + h * C + c + 3, alpha * u.w);
        }
    }
}

__global__ void elu_kernel(float* p, int n) {
    int i = blockIdx.x * blockDim.x + threadIdx.x;
    if (i < n) {
        float x = p[i];
        p[i] = x > 0.f ? x : expm1f(x);
    }
}

// classifier head: logits = tanh(pooled@Wb + bb)@Wc + bc
__global__ void head_kernel(const float* __restrict__ pooled, const float* __restrict__ Wb,
                            const float* __restrict__ bb, const float* __restrict__ Wc,
                            const float* __restrict__ bc, float* __restrict__ out) {
    __shared__ float t[16];
    int row = blockIdx.x;
    int j = threadIdx.x;
    if (j < 16) {
        float acc = bb[j];
        for (int k = 0; k < 768; ++k) acc += pooled[(size_t)row * 768 + k] * Wb[(size_t)k * 16 + j];
        t[j] = tanhf(acc);
    }
    __syncthreads();
    if (j < 4) {
        float acc = bc[j];
#pragma unroll
        for (int i = 0; i < 16; ++i) acc += t[i] * Wc[i * 4 + j];
        out[(size_t)row * 4 + j] = acc;
    }
}

// ---------------------------------------------------------------------------
extern "C" void kernel_launch(void* const* d_in, const int* in_sizes, int n_in,
                              void* d_out, int out_size, void* d_ws, size_t ws_size,
                              hipStream_t stream) {
    const float* x      = (const float*)d_in[0];
    const float* pooled = (const float*)d_in[1];
    const int*   esrc   = (const int*)d_in[2];
    const int*   edst   = (const int*)d_in[3];
    const float* Wl1    = (const float*)d_in[4];
    const float* Wr1    = (const float*)d_in[5];
    const float* att1   = (const float*)d_in[6];
    const float* b1     = (const float*)d_in[7];
    const float* Wl2    = (const float*)d_in[8];
    const float* Wr2    = (const float*)d_in[9];
    const float* att2   = (const float*)d_in[10];
    const float* b2     = (const float*)d_in[11];
    const float* Wb     = (const float*)d_in[12];
    const float* bb     = (const float*)d_in[13];
    const float* Wc     = (const float*)d_in[14];
    const float* bc     = (const float*)d_in[15];

    const int NN = in_sizes[0] / 768;   // 50000 nodes
    const int E  = in_sizes[2];         // 800000 edges
    const int B  = in_sizes[1] / 768;   // 64 pooled rows
    const int ET = E + NN;              // with self-loops

    // workspace carve-up (256B aligned)
    char* base = (char*)d_ws;
    size_t off = 0;
    auto carve = [&](size_t bytes) -> char* {
        char* p = base + off;
        off = (off + bytes + 255) & ~(size_t)255;
        return p;
    };
    int*      Bp1    = (int*)carve((size_t)8 * 24 * 256 * 4);       // packed Wl1|Wr1
    int*      Bp2    = (int*)carve((size_t)2 * 2 * 256 * 4);        // packed Wl2|Wr2
    float*    xlr1   = (float*)carve((size_t)NN * 128 * 4);         // [xl1 | xr1]
    float*    hbuf   = (float*)carve((size_t)NN * 64 * 4);          // layer-1 output
    float*    xlr2   = (float*)carve((size_t)NN * 32 * 4);          // [xl2 | xr2]
    float*    score1 = (float*)carve((size_t)ET * 4 * 4);
    uint32_t* smax1  = (uint32_t*)carve((size_t)NN * 4 * 4);
    float*    den1   = (float*)carve((size_t)NN * 4 * 4);
    float*    score2 = (float*)carve((size_t)ET * 4);
    uint32_t* smax2  = (uint32_t*)carve((size_t)NN * 4);
    float*    den2   = (float*)carve((size_t)NN * 4);

    float* hout   = (float*)d_out;            // [NN,16]
    float* logits = (float*)d_out + (size_t)NN * 16;

    const int TB = 256;
    auto gsz = [&](int n) { return (unsigned)((n + TB - 1) / TB); };

    // ---- layer 1 GEMM: xlr1 = x @ [Wl1|Wr1]  (bf16 WMMA, f32 accumulate) ----
    pack_b_kernel<<<gsz(4 * 24 * 256), TB, 0, stream>>>(Wl1, 768, 64, Bp1, 0, 24);
    pack_b_kernel<<<gsz(4 * 24 * 256), TB, 0, stream>>>(Wr1, 768, 64, Bp1, 4, 24);
    {
        int mtiles = (NN + 15) / 16;
        gemm_bf16_kernel<8><<<(unsigned)((mtiles + 7) / 8), TB, 0, stream>>>(
            x, Bp1, xlr1, NN, 768, 768, 128);
    }

    // ---- layer 1 edge softmax + aggregate (bias folded into init) ----
    init_u32_kernel<<<gsz(NN * 4), TB, 0, stream>>>(smax1, NN * 4, 0u);
    init_f32_kernel<<<gsz(NN * 4), TB, 0, stream>>>(den1, NN * 4, 0.f);
    init_bias_kernel<<<gsz(NN * 64), TB, 0, stream>>>(hbuf, NN, 64, b1);
    edge_score_kernel<4, 16><<<gsz(ET), TB, 0, stream>>>(
        xlr1, xlr1 + 64, 128, att1, esrc, edst, E, NN, score1, smax1);
    edge_exp_kernel<4><<<gsz(ET), TB, 0, stream>>>(edst, E, NN, score1, smax1, den1);
    edge_aggr_kernel<4, 16><<<gsz(ET), TB, 0, stream>>>(
        xlr1, 128, esrc, edst, E, NN, score1, den1, hbuf, 64);
    elu_kernel<<<gsz(NN * 64), TB, 0, stream>>>(hbuf, NN * 64);

    // ---- layer 2 GEMM: xlr2 = h @ [Wl2|Wr2] ----
    pack_b_kernel<<<gsz(1 * 2 * 256), TB, 0, stream>>>(Wl2, 64, 16, Bp2, 0, 2);
    pack_b_kernel<<<gsz(1 * 2 * 256), TB, 0, stream>>>(Wr2, 64, 16, Bp2, 1, 2);
    {
        int mtiles = (NN + 15) / 16;
        gemm_bf16_kernel<2><<<(unsigned)((mtiles + 7) / 8), TB, 0, stream>>>(
            hbuf, Bp2, xlr2, NN, 64, 64, 32);
    }

    // ---- layer 2 edge softmax + aggregate directly into d_out ----
    init_u32_kernel<<<gsz(NN), TB, 0, stream>>>(smax2, NN, 0u);
    init_f32_kernel<<<gsz(NN), TB, 0, stream>>>(den2, NN, 0.f);
    init_bias_kernel<<<gsz(NN * 16), TB, 0, stream>>>(hout, NN, 16, b2);
    edge_score_kernel<1, 16><<<gsz(ET), TB, 0, stream>>>(
        xlr2, xlr2 + 16, 32, att2, esrc, edst, E, NN, score2, smax2);
    edge_exp_kernel<1><<<gsz(ET), TB, 0, stream>>>(edst, E, NN, score2, smax2, den2);
    edge_aggr_kernel<1, 16><<<gsz(ET), TB, 0, stream>>>(
        xlr2, 32, esrc, edst, E, NN, score2, den2, hout, 16);

    // ---- classifier head ----
    head_kernel<<<(unsigned)B, 32, 0, stream>>>(pooled, Wb, bb, Wc, bc, logits);
}
